// Attention_layer_32762010534057
// MI455X (gfx1250) — matrix operand order repaired and checked by reading, not compiled, not run on hardware
//
#include <hip/hip_runtime.h>
#include <math.h>

// ---- problem dims ----
#define CH   256   // input channels
#define SDIM 128   // shrink channels
#define DD   64    // depth
#define HWN  256   // H*W

typedef __bf16 bf16;
typedef __attribute__((ext_vector_type(16))) __bf16 v16bf;
typedef __attribute__((ext_vector_type(8)))  __bf16 v8bf;
typedef __attribute__((ext_vector_type(8)))  float  v8f;

// ---- LDS layout (bytes); pitches 16B-aligned, bank-skewed ----
#define XB_P 264      // bf16 pitch (row = one d, cols = c)
#define KT_P 136      // bf16 pitch (row = one d, cols = s)  [also attT]
#define VS_P 72       // bf16 pitch (row = one s, cols = d)
#define ST_P 68       // f32 pitch (row = one j, cols = i; cols 64..67 = partial max)
#define PT_P 72       // bf16 pitch (row = one j, cols = i)
#define OFF_XB 0       // 64*264*2  = 33792
#define OFF_KT 33792   // 64*136*2  = 17408
#define OFF_QT 51200   // 17408
#define OFF_VS 68608   // 128*72*2  = 18432
#define OFF_ST 87040   // 64*68*4   = 17408
#define OFF_PT 104448  // 64*72*2   = 9216
#define OFF_SC 113664  // 64*4*4    = 1024 (softmax partial sums)
#define SMEM_BYTES 114688

// bf16 weights staged in d_ws (elements)
#define WKB_OFF 0
#define WQB_OFF 32768
#define WVB_OFF 65536
#define WOB_OFF 98304

// ---------- fragment helpers (wave32 WMMA operands, contiguous k order used in this file) ----------
// A (16x32 bf16, row-major [M][K]): lane m = base+(lane&15), half h=lane>>4:
//   elements 0..15 = A[m][k0+16h .. +15]
__device__ __forceinline__ v16bf ld_a_bf(const bf16* __restrict__ rowp, int k0, int hh) {
    v8bf lo = *(const v8bf*)(rowp + k0 + 16 * hh);
    v8bf hi = *(const v8bf*)(rowp + k0 + 16 * hh + 8);
    return __builtin_shufflevector(lo, hi, 0,1,2,3,4,5,6,7,8,9,10,11,12,13,14,15);
}
// B (32x16 bf16, [K][N]) from K-contiguous storage (B^T rows):
//   lane n = base+(lane&15) reads 16 contiguous K values at k0+16h
__device__ __forceinline__ v16bf ld_b_bf(const bf16* __restrict__ p) {
    v8bf lo = *(const v8bf*)(p);
    v8bf hi = *(const v8bf*)(p + 8);
    return __builtin_shufflevector(lo, hi, 0,1,2,3,4,5,6,7,8,9,10,11,12,13,14,15);
}
__device__ __forceinline__ v8f wmma_bf16(v16bf a, v16bf b, v8f c) {
    return __builtin_amdgcn_wmma_f32_16x16x32_bf16(false, a, false, b, (short)0, c, false, false);
}
__device__ __forceinline__ v8f zero8() { v8f z = {}; return z; }

// ---- one-shot f32 -> bf16 weight conversion into d_ws ----
__global__ void __launch_bounds__(256)
cvt_weights_kernel(const float* __restrict__ wk, const float* __restrict__ wq,
                   const float* __restrict__ wv, const float* __restrict__ wo,
                   bf16* __restrict__ wb)
{
    int i = blockIdx.x * 256 + threadIdx.x;   // 0..32767
    wb[WKB_OFF + i] = (bf16)wk[i];
    wb[WQB_OFF + i] = (bf16)wq[i];
    wb[WVB_OFF + i] = (bf16)wv[i];
    wb[WOB_OFF + i] = (bf16)wo[i];
}

// One workgroup per (b, h*w). 8 wave32 per workgroup.
__global__ void __launch_bounds__(256)
attn3d_fused_kernel(const float* __restrict__ x,
                    const bf16* __restrict__ wkb, const float* __restrict__ bk,
                    const bf16* __restrict__ wqb, const float* __restrict__ bq,
                    const bf16* __restrict__ wvb, const float* __restrict__ bv,
                    const bf16* __restrict__ wob, const float* __restrict__ bo,
                    float* __restrict__ out)
{
    extern __shared__ char smem[];
    bf16*  XB = (bf16*) (smem + OFF_XB);   // [DD][XB_P]  x^T, bf16
    bf16*  KT = (bf16*) (smem + OFF_KT);   // [DD][KT_P]  k^T
    bf16*  QT = (bf16*) (smem + OFF_QT);   // [DD][KT_P]  q^T
    bf16*  VS = (bf16*) (smem + OFF_VS);   // [SDIM][VS_P] v
    float* ST = (float*)(smem + OFF_ST);   // [DD][ST_P]  scores^T (j-major)
    bf16*  PT = (bf16*) (smem + OFF_PT);   // [DD][PT_P]  probs^T
    float* SC = (float*)(smem + OFF_SC);   // [DD][4]     softmax partial sums
    bf16*  AT = (bf16*) (smem + OFF_KT);   // [DD][KT_P]  att^T (reuses KT)

    const int tid  = threadIdx.x;
    const int wave = tid >> 5;
    const int lane = tid & 31;
    const int l15  = lane & 15;
    const int hh   = lane >> 4;

    const int b  = blockIdx.x >> 8;
    const int hw = blockIdx.x & 255;
    const size_t xbase = (size_t)b * (CH * DD * HWN) + hw;

    // ---- phase 0: x[b, :, :, hw] -> XB[d][c] (bf16, converted once) ----
    for (int idx = tid; idx < CH * DD; idx += 256) {
        int c = idx >> 6, d = idx & 63;
        XB[d * XB_P + c] = (bf16)x[xbase + (size_t)c * (DD * HWN) + (size_t)d * HWN];
    }
    __syncthreads();

    // ---- phase 1: k,q,v projections.  M=SDIM strip per wave, N=DD, K=CH ----
    {
        v8f acc[3][4];
        #pragma unroll
        for (int p = 0; p < 3; ++p)
            #pragma unroll
            for (int nt = 0; nt < 4; ++nt) acc[p][nt] = zero8();

        const int mrow = wave * 16 + l15;
        for (int ks = 0; ks < CH; ks += 32) {
            v16bf Ak = ld_a_bf(wkb + mrow * CH, ks, hh);   // global bf16, L2-resident
            v16bf Aq = ld_a_bf(wqb + mrow * CH, ks, hh);
            v16bf Av = ld_a_bf(wvb + mrow * CH, ks, hh);
            #pragma unroll
            for (int nt = 0; nt < 4; ++nt) {
                v16bf Bx = ld_b_bf(XB + (nt * 16 + l15) * XB_P + ks + 16 * hh);
                acc[0][nt] = wmma_bf16(Ak, Bx, acc[0][nt]);
                acc[1][nt] = wmma_bf16(Aq, Bx, acc[1][nt]);
                acc[2][nt] = wmma_bf16(Av, Bx, acc[2][nt]);
            }
        }
        float bkr[8], bqr[8], bvr[8];
        #pragma unroll
        for (int e = 0; e < 8; ++e) {
            int s = wave * 16 + 8 * hh + e;
            bkr[e] = bk[s]; bqr[e] = bq[s]; bvr[e] = bv[s];
        }
        #pragma unroll
        for (int nt = 0; nt < 4; ++nt) {
            int dd = nt * 16 + l15;
            #pragma unroll
            for (int e = 0; e < 8; ++e) {
                int s = wave * 16 + 8 * hh + e;
                KT[dd * KT_P + s] = (bf16)(acc[0][nt][e] + bkr[e]);
                QT[dd * KT_P + s] = (bf16)(acc[1][nt][e] + bqr[e]);
                VS[s * VS_P + dd] = (bf16)(acc[2][nt][e] + bvr[e]);
            }
        }
    }
    __syncthreads();

    // ---- phase 2: scores[i][j] = sum_s k[s][i]*q[s][j] * scale ----
    {
        const int it = wave >> 1;
        const float scale = 0.088388347648318447f; // 1/sqrt(128)
        #pragma unroll
        for (int jj = 0; jj < 2; ++jj) {
            int jt = (wave & 1) * 2 + jj;
            v8f sc = zero8();
            const bf16* arow = KT + (it * 16 + l15) * KT_P;   // A[i][s]
            const bf16* brow = QT + (jt * 16 + l15) * KT_P;   // B col j contiguous in s
            for (int ks = 0; ks < SDIM; ks += 32) {
                v16bf A  = ld_a_bf(arow, ks, hh);
                v16bf Bq = ld_b_bf(brow + ks + 16 * hh);
                sc = wmma_bf16(A, Bq, sc);
            }
            #pragma unroll
            for (int e = 0; e < 8; ++e) {
                int i = it * 16 + 8 * hh + e;
                int j = jt * 16 + l15;
                ST[j * ST_P + i] = sc[e] * scale;
            }
        }
    }
    __syncthreads();

    // ---- phase 3: softmax over i, 4 threads per row j ----
    {
        const int j = tid >> 2, q = tid & 3;
        const float* srow = ST + j * ST_P + q * 16;
        float m = srow[0];
        #pragma unroll
        for (int i = 1; i < 16; ++i) m = fmaxf(m, srow[i]);
        ST[j * ST_P + 64 + q] = m;                       // partial max in spare cols
        __syncthreads();
        m = fmaxf(fmaxf(ST[j * ST_P + 64], ST[j * ST_P + 65]),
                  fmaxf(ST[j * ST_P + 66], ST[j * ST_P + 67]));
        float s = 0.f;
        #pragma unroll
        for (int i = 0; i < 16; ++i) s += __expf(srow[i] - m);
        SC[j * 4 + q] = s;
        __syncthreads();
        float inv = 1.f / (SC[j * 4 + 0] + SC[j * 4 + 1] + SC[j * 4 + 2] + SC[j * 4 + 3]);
        bf16* prow = PT + j * PT_P + q * 16;
        #pragma unroll
        for (int i = 0; i < 16; ++i) prow[i] = (bf16)(__expf(srow[i] - m) * inv);
    }
    __syncthreads();

    // ---- phase 4: att[s][j] = sum_i v[s][i]*a[i][j]  (M=SDIM,N=DD,K=DD) ----
    {
        v8f acc[4];
        #pragma unroll
        for (int nt = 0; nt < 4; ++nt) acc[nt] = zero8();
        const bf16* arow = VS + (wave * 16 + l15) * VS_P;     // A[s][i]
        #pragma unroll
        for (int ks = 0; ks < 64; ks += 32) {
            v16bf A = ld_a_bf(arow, ks, hh);
            #pragma unroll
            for (int nt = 0; nt < 4; ++nt) {
                v16bf Bp = ld_b_bf(PT + (nt * 16 + l15) * PT_P + ks + 16 * hh);
                acc[nt] = wmma_bf16(A, Bp, acc[nt]);
            }
        }
        #pragma unroll
        for (int nt = 0; nt < 4; ++nt) {
            int j = nt * 16 + l15;
            #pragma unroll
            for (int e = 0; e < 8; ++e) {
                int s = wave * 16 + 8 * hh + e;
                AT[j * KT_P + s] = (bf16)acc[nt][e];   // KT region dead now
            }
        }
    }
    __syncthreads();

    // ---- phase 5: out[c][d] = wo x att + bo + x  (M=CH,N=DD,K=SDIM) ----
    for (int msi = 0; msi < 2; ++msi) {
        const int ms = wave * 2 + msi;
        const int crow = ms * 16 + l15;
        v8f acc[4];
        #pragma unroll
        for (int nt = 0; nt < 4; ++nt) acc[nt] = zero8();
        for (int ks = 0; ks < SDIM; ks += 32) {
            v16bf A = ld_a_bf(wob + crow * SDIM, ks, hh);
            #pragma unroll
            for (int nt = 0; nt < 4; ++nt) {
                v16bf Bt = ld_b_bf(AT + (nt * 16 + l15) * KT_P + ks + 16 * hh);
                acc[nt] = wmma_bf16(A, Bt, acc[nt]);
            }
        }
        float bor[8];
        #pragma unroll
        for (int e = 0; e < 8; ++e) bor[e] = bo[ms * 16 + 8 * hh + e];
        #pragma unroll
        for (int nt = 0; nt < 4; ++nt) {
            int d = nt * 16 + l15;
            #pragma unroll
            for (int e = 0; e < 8; ++e) {
                int c = ms * 16 + 8 * hh + e;
                size_t gidx = xbase + (size_t)c * (DD * HWN) + (size_t)d * HWN;
                out[gidx] = acc[nt][e] + bor[e] + x[gidx];   // fp32 residual (L2 hit)
            }
        }
    }
}

extern "C" void kernel_launch(void* const* d_in, const int* in_sizes, int n_in,
                              void* d_out, int out_size, void* d_ws, size_t ws_size,
                              hipStream_t stream) {
    (void)in_sizes; (void)n_in; (void)out_size; (void)ws_size;
    const float* x  = (const float*)d_in[0];
    const float* wk = (const float*)d_in[1];
    const float* bk = (const float*)d_in[2];
    const float* wq = (const float*)d_in[3];
    const float* bq = (const float*)d_in[4];
    const float* wv = (const float*)d_in[5];
    const float* bv = (const float*)d_in[6];
    const float* wo = (const float*)d_in[7];
    const float* bo = (const float*)d_in[8];
    float* out = (float*)d_out;
    bf16* wb = (bf16*)d_ws;   // 262144 bytes of scratch used

    // stage bf16 weights (L2-resident for the main kernel)
    cvt_weights_kernel<<<dim3(128), dim3(256), 0, stream>>>(wk, wq, wv, wo, wb);

    dim3 grid(8 * 256);   // one block per (b, h*w)
    dim3 block(256);      // 8 wave32
    attn3d_fused_kernel<<<grid, block, SMEM_BYTES, stream>>>(
        x, wb + WKB_OFF, bk, wb + WQB_OFF, bq, wb + WVB_OFF, bv, wb + WOB_OFF, bo, out);
}
